// MultiHeadAttention_83339545411608
// MI455X (gfx1250) — compile-verified, matrix operand order checked by reading, not run on hardware
//
#include <hip/hip_runtime.h>
#include <hip/hip_bf16.h>

// ---------------------------------------------------------------------------
// MultiHeadAttention layer for MI455X (gfx1250), bf16 WMMA path.
// B=4, L=1024, D=1024, H=16, DK=DV=64.
// Outputs (concatenated in d_out): y [B,L,D] f32, attn [H*B,L,L] f32.
// ---------------------------------------------------------------------------

#define BB 4
#define LL 1024
#define DD 1024
#define HH 16
#define DK 64
#define DV 64
#define MM (BB * LL)        // 4096 rows for the big GEMMs
#define KK DD               // 1024
#define NN (HH * DK)        // 1024

typedef __attribute__((ext_vector_type(16))) __bf16 v16bf;
typedef __attribute__((ext_vector_type(8)))  float  v8f;

union BFrag {
    v16bf v;
    uint4 q[2];
};

// round-to-nearest-even f32 -> bf16 bits
__device__ __forceinline__ unsigned bfb(float x) {
    unsigned u = __float_as_uint(x);
    unsigned r = u + 0x7FFFu + ((u >> 16) & 1u);
    return r >> 16;
}
__device__ __forceinline__ unsigned pk2(float a, float b) {
    return (bfb(b) << 16) | (bfb(a) & 0xFFFFu);
}

// 16-bit A-matrix 16x32 fragment (also used for B as Bt rows == columns):
// lanes 0-15: row = lane,    K = [0..7]  and [16..23]
// lanes 16-31: row = lane-16, K = [8..15] and [24..31]
__device__ __forceinline__ void load_frag_bf16(BFrag& f, const unsigned short* base,
                                               int ld, int lane) {
    int half = (lane >> 4) & 1;
    int r    = lane & 15;
    const unsigned short* p = base + (size_t)r * ld + half * 8;
    f.q[0] = *(const uint4*)(p);
    f.q[1] = *(const uint4*)(p + 16);
}

// same fragment but source is f32; convert to bf16 (RNE) in-register
__device__ __forceinline__ void load_frag_f32(BFrag& f, const float* base,
                                              int ld, int lane) {
    int half = (lane >> 4) & 1;
    int r    = lane & 15;
    const float* p = base + (size_t)r * ld + half * 8;
    float4 a0 = *(const float4*)(p);
    float4 a1 = *(const float4*)(p + 4);
    float4 a2 = *(const float4*)(p + 16);
    float4 a3 = *(const float4*)(p + 20);
    f.q[0] = make_uint4(pk2(a0.x, a0.y), pk2(a0.z, a0.w), pk2(a1.x, a1.y), pk2(a1.z, a1.w));
    f.q[1] = make_uint4(pk2(a2.x, a2.y), pk2(a2.z, a2.w), pk2(a3.x, a3.y), pk2(a3.z, a3.w));
}

__device__ __forceinline__ v8f wmma_bf16(const BFrag& a, const BFrag& b, v8f c) {
    return __builtin_amdgcn_wmma_f32_16x16x32_bf16(false, a.v, false, b.v,
                                                   (short)0, c, false, false);
}

// ---------------------------------------------------------------------------
// Converters
// ---------------------------------------------------------------------------

// flat f32 -> bf16, 4 elements / thread
__global__ __launch_bounds__(256) void cvt_bf16_kernel(const float* __restrict__ src,
                                                       unsigned short* __restrict__ dst,
                                                       int n4) {
    int i = blockIdx.x * blockDim.x + threadIdx.x;
    if (i >= n4) return;
    float4 v = ((const float4*)src)[i];
    ((uint2*)dst)[i] = make_uint2(pk2(v.x, v.y), pk2(v.z, v.w));
}

// WT[n,k] = bf16(W[k,n]); rows=K, cols=N, both 1024. 32x32 LDS tile transpose.
__global__ __launch_bounds__(256) void transpose_cvt_kernel(const float* __restrict__ W,
                                                            unsigned short* __restrict__ WT,
                                                            int rows, int cols) {
    __shared__ float t[32][33];
    int bx = blockIdx.x * 32;   // col base (n)
    int by = blockIdx.y * 32;   // row base (k)
    int x = threadIdx.x, y = threadIdx.y;   // block (32, 8)
#pragma unroll
    for (int j = 0; j < 4; ++j)
        t[y + j * 8][x] = W[(size_t)(by + y + j * 8) * cols + bx + x];
    __syncthreads();
#pragma unroll
    for (int j = 0; j < 4; ++j)
        WT[(size_t)(bx + y + j * 8) * rows + by + x] = (unsigned short)bfb(t[x][y + j * 8]);
}

// ---------------------------------------------------------------------------
// Generic bf16 WMMA GEMM: C[M,N] = A[M,K] * Bt[N,K]^T + bias
// wave: 16(M) x 64(N) tile, 4 accumulators; WG 8 waves = 128 x 64.
// Software-pipelined k-loop (ping-pong fragments) so loads for k+32 are in
// flight while WMMAs for k execute.
// mode 0: bf16 row-major out     (qh, kh)
// mode 1: bf16 per-head transposed vhT[(b*H+h)*DV + d][l]
// mode 2: f32 out = acc + bias + residual   (FC + residual)
// ---------------------------------------------------------------------------
__global__ __launch_bounds__(256) void gemm_bf16_kernel(const unsigned short* __restrict__ A,
                                                        const unsigned short* __restrict__ Bt,
                                                        const float* __restrict__ bias,
                                                        const float* __restrict__ residual,
                                                        void* __restrict__ out,
                                                        int M, int N, int K, int mode) {
    int tid  = threadIdx.x;
    int wave = tid >> 5;
    int lane = tid & 31;
    int half = (lane >> 4) & 1;
    int l15  = lane & 15;

    int m0     = blockIdx.y * 128 + wave * 16;
    int n_base = blockIdx.x * 64;
    if (m0 >= M) return;

    v8f acc[4] = {};
    const unsigned short* Abase = A + (size_t)m0 * K;

    BFrag a0, a1, b0[4], b1[4];
    load_frag_bf16(a0, Abase, K, lane);
#pragma unroll
    for (int t = 0; t < 4; ++t)
        load_frag_bf16(b0[t], Bt + (size_t)(n_base + t * 16) * K, K, lane);

    for (int k0 = 0; k0 < K; k0 += 64) {
        int kn = k0 + 32;
        __builtin_prefetch(Abase + k0 + 128, 0, 0);
        load_frag_bf16(a1, Abase + kn, K, lane);
#pragma unroll
        for (int t = 0; t < 4; ++t)
            load_frag_bf16(b1[t], Bt + (size_t)(n_base + t * 16) * K + kn, K, lane);
#pragma unroll
        for (int t = 0; t < 4; ++t)
            acc[t] = wmma_bf16(a0, b0[t], acc[t]);

        int kn2 = (k0 + 64 < K) ? k0 + 64 : 0;   // clamp: last prefetch unused
        load_frag_bf16(a0, Abase + kn2, K, lane);
#pragma unroll
        for (int t = 0; t < 4; ++t)
            load_frag_bf16(b0[t], Bt + (size_t)(n_base + t * 16) * K + kn2, K, lane);
#pragma unroll
        for (int t = 0; t < 4; ++t)
            acc[t] = wmma_bf16(a1, b1[t], acc[t]);
    }

#pragma unroll
    for (int t = 0; t < 4; ++t) {
        int n0 = n_base + t * 16;
#pragma unroll
        for (int i = 0; i < 8; ++i) {
            int r = m0 + half * 8 + i;       // global M
            int c = n0 + l15;                // global N
            float val = acc[t][i] + bias[c];
            if (mode == 0) {
                ((unsigned short*)out)[(size_t)r * N + c] = (unsigned short)bfb(val);
            } else if (mode == 1) {
                int b = r >> 10, l = r & 1023;
                int h = c >> 6,  d = c & 63;
                ((unsigned short*)out)[(((size_t)(b * HH + h) * DV + d) << 10) + l] =
                    (unsigned short)bfb(val);
            } else {
                size_t idx = (size_t)r * N + c;
                ((float*)out)[idx] = val + residual[idx];
            }
        }
    }
}

// ---------------------------------------------------------------------------
// Fused scores + mask + softmax.
// Grid: (L/16, H, B). WG: 256 threads, 8 waves. Each wave computes the
// 16(q) x 128(k) strip of scores (WMMA over DK=64), all strips staged in
// 64KB LDS, then wave32 shfl softmax, attn written once as [H,B,L,L] f32.
// ---------------------------------------------------------------------------
__global__ __launch_bounds__(256) void attn_scores_softmax_kernel(
        const unsigned short* __restrict__ qh,   // [B*L, H*DK] bf16
        const unsigned short* __restrict__ kh,   // [B*L, H*DK] bf16
        const unsigned char*  __restrict__ mask, // [B, L, L] bool
        float* __restrict__ attn) {              // [H, B, L, L] f32
    __shared__ float sc[16][LL];                 // 64 KB

    int tid  = threadIdx.x;
    int wave = tid >> 5;
    int lane = tid & 31;
    int half = (lane >> 4) & 1;
    int l15  = lane & 15;

    int qtile = blockIdx.x;
    int h     = blockIdx.y;
    int b     = blockIdx.z;

    const unsigned short* qbase = qh + ((size_t)(b * LL + qtile * 16) * NN) + h * DK;
    const unsigned short* kbase = kh + ((size_t)b * LL) * NN + h * DK;

    v8f acc[8] = {};
#pragma unroll
    for (int k0 = 0; k0 < DK; k0 += 32) {
        BFrag a;
        load_frag_bf16(a, qbase + k0, NN, lane);
#pragma unroll
        for (int t = 0; t < 8; ++t) {
            BFrag kb;
            load_frag_bf16(kb, kbase + (size_t)(wave * 128 + t * 16) * NN + k0, NN, lane);
            acc[t] = wmma_bf16(a, kb, acc[t]);
        }
    }

    const float scale = 0.125f;  // 1/sqrt(64)
#pragma unroll
    for (int t = 0; t < 8; ++t)
#pragma unroll
        for (int i = 0; i < 8; ++i)
            sc[half * 8 + i][wave * 128 + t * 16 + l15] = acc[t][i] * scale;
    __syncthreads();

    // softmax: 2 rows per wave
#pragma unroll
    for (int rr = 0; rr < 2; ++rr) {
        int r  = wave * 2 + rr;
        int qg = qtile * 16 + r;
        const unsigned char* mrow = mask + ((size_t)b * LL + qg) * LL;

        float m = -3.0e38f;
#pragma unroll
        for (int i = 0; i < 32; ++i) {
            int c = lane + 32 * i;
            float s = sc[r][c];
            if (mrow[c]) s = -1.0e9f;
            sc[r][c] = s;
            m = fmaxf(m, s);
        }
#pragma unroll
        for (int off = 16; off >= 1; off >>= 1)
            m = fmaxf(m, __shfl_xor(m, off, 32));

        float sum = 0.0f;
#pragma unroll
        for (int i = 0; i < 32; ++i) {
            int c = lane + 32 * i;
            float e = __expf(sc[r][c] - m);
            sc[r][c] = e;
            sum += e;
        }
#pragma unroll
        for (int off = 16; off >= 1; off >>= 1)
            sum += __shfl_xor(sum, off, 32);
        float inv = 1.0f / sum;

        float* orow = attn + (((size_t)(h * BB + b) * LL) + qg) * LL;
#pragma unroll
        for (int i = 0; i < 32; ++i) {
            int c = lane + 32 * i;
            orow[c] = sc[r][c] * inv;
        }
    }
}

// ---------------------------------------------------------------------------
// ctx[b*L+q, h*DV+d] = sum_k attn[h,b,q,k] * vhT[(b*H+h)*DV+d, k]
// Grid: (L/128, B*H). Wave: 16(q) x 64(d), k-loop software-pipelined.
// attn read f32 -> bf16 fragments in-register.
// ---------------------------------------------------------------------------
__global__ __launch_bounds__(256) void attn_av_kernel(const float* __restrict__ attn,
                                                      const unsigned short* __restrict__ vhT,
                                                      unsigned short* __restrict__ ctx) {
    int tid  = threadIdx.x;
    int wave = tid >> 5;
    int lane = tid & 31;
    int half = (lane >> 4) & 1;
    int l15  = lane & 15;

    int bh = blockIdx.y;
    int b  = bh >> 4;
    int h  = bh & 15;
    int q0 = blockIdx.x * 128 + wave * 16;

    const float* arow = attn + (((size_t)(h * BB + b) * LL) + q0) * LL;
    const unsigned short* vbase = vhT + ((size_t)(b * HH + h) * DV) * LL;

    v8f acc[4] = {};
    BFrag a0, a1, v0[4], v1[4];
    load_frag_f32(a0, arow, LL, lane);
#pragma unroll
    for (int t = 0; t < 4; ++t)
        load_frag_bf16(v0[t], vbase + (size_t)(t * 16) * LL, LL, lane);

    for (int k0 = 0; k0 < LL; k0 += 64) {
        int kn = k0 + 32;
        __builtin_prefetch(arow + k0 + 128, 0, 0);
        load_frag_f32(a1, arow + kn, LL, lane);
#pragma unroll
        for (int t = 0; t < 4; ++t)
            load_frag_bf16(v1[t], vbase + (size_t)(t * 16) * LL + kn, LL, lane);
#pragma unroll
        for (int t = 0; t < 4; ++t)
            acc[t] = wmma_bf16(a0, v0[t], acc[t]);

        int kn2 = (k0 + 64 < LL) ? k0 + 64 : 0;   // clamp: last prefetch unused
        load_frag_f32(a0, arow + kn2, LL, lane);
#pragma unroll
        for (int t = 0; t < 4; ++t)
            load_frag_bf16(v0[t], vbase + (size_t)(t * 16) * LL + kn2, LL, lane);
#pragma unroll
        for (int t = 0; t < 4; ++t)
            acc[t] = wmma_bf16(a1, v1[t], acc[t]);
    }

#pragma unroll
    for (int t = 0; t < 4; ++t)
#pragma unroll
        for (int i = 0; i < 8; ++i) {
            int q = q0 + half * 8 + i;
            int d = t * 16 + l15;
            ctx[(size_t)(b * LL + q) * NN + h * DV + d] = (unsigned short)bfb(acc[t][i]);
        }
}

// ---------------------------------------------------------------------------
// LayerNorm: one wave per 1024-element row (values kept in registers).
// ---------------------------------------------------------------------------
__global__ __launch_bounds__(256) void layernorm_kernel(const float* __restrict__ x,
                                                        const float* __restrict__ gamma,
                                                        const float* __restrict__ beta,
                                                        float* __restrict__ y) {
    int wave = threadIdx.x >> 5;
    int lane = threadIdx.x & 31;
    int row  = blockIdx.x * 8 + wave;

    const float4* xr = (const float4*)(x + (size_t)row * DD);
    float4 v[8];
    float sum = 0.0f;
#pragma unroll
    for (int i = 0; i < 8; ++i) {
        v[i] = xr[lane + 32 * i];
        sum += v[i].x + v[i].y + v[i].z + v[i].w;
    }
#pragma unroll
    for (int off = 16; off >= 1; off >>= 1)
        sum += __shfl_xor(sum, off, 32);
    float mu = sum * (1.0f / DD);

    float vs = 0.0f;
#pragma unroll
    for (int i = 0; i < 8; ++i) {
        float dx0 = v[i].x - mu, dx1 = v[i].y - mu, dx2 = v[i].z - mu, dx3 = v[i].w - mu;
        vs += dx0 * dx0 + dx1 * dx1 + dx2 * dx2 + dx3 * dx3;
    }
#pragma unroll
    for (int off = 16; off >= 1; off >>= 1)
        vs += __shfl_xor(vs, off, 32);
    float inv = rsqrtf(vs * (1.0f / DD) + 1e-5f);

    float4* yr = (float4*)(y + (size_t)row * DD);
    const float4* g4 = (const float4*)gamma;
    const float4* b4 = (const float4*)beta;
#pragma unroll
    for (int i = 0; i < 8; ++i) {
        int j = lane + 32 * i;
        float4 g = g4[j], bb = b4[j];
        float4 o;
        o.x = (v[i].x - mu) * inv * g.x + bb.x;
        o.y = (v[i].y - mu) * inv * g.y + bb.y;
        o.z = (v[i].z - mu) * inv * g.z + bb.z;
        o.w = (v[i].w - mu) * inv * g.w + bb.w;
        yr[j] = o;
    }
}

// ---------------------------------------------------------------------------
// Launch
// ---------------------------------------------------------------------------
extern "C" void kernel_launch(void* const* d_in, const int* in_sizes, int n_in,
                              void* d_out, int out_size, void* d_ws, size_t ws_size,
                              hipStream_t stream) {
    (void)in_sizes; (void)n_in; (void)out_size; (void)ws_size;

    const float* q    = (const float*)d_in[0];
    const float* k    = (const float*)d_in[1];
    const float* v    = (const float*)d_in[2];
    const unsigned char* mask = (const unsigned char*)d_in[3];
    const float* wq   = (const float*)d_in[4];
    const float* bq   = (const float*)d_in[5];
    const float* wk   = (const float*)d_in[6];
    const float* bk   = (const float*)d_in[7];
    const float* wv   = (const float*)d_in[8];
    const float* bv   = (const float*)d_in[9];
    const float* wfc  = (const float*)d_in[10];
    const float* bfc  = (const float*)d_in[11];
    const float* gamma = (const float*)d_in[12];
    const float* beta  = (const float*)d_in[13];

    float* y_out    = (float*)d_out;                        // [B,L,D]
    float* attn_out = y_out + (size_t)BB * LL * DD;         // [H,B,L,L]

    // workspace layout (bytes)
    char* ws = (char*)d_ws;
    const size_t WSZ = (size_t)DD * NN * 2;                 // 2 MB per bf16 weight
    const size_t XSZ = (size_t)MM * KK * 2;                 // 8 MB per bf16 activation
    unsigned short* wqT  = (unsigned short*)(ws + 0 * WSZ);
    unsigned short* wkT  = (unsigned short*)(ws + 1 * WSZ);
    unsigned short* wvT  = (unsigned short*)(ws + 2 * WSZ);
    unsigned short* wfcT = (unsigned short*)(ws + 3 * WSZ);
    char* p = ws + 4 * WSZ;
    unsigned short* xq   = (unsigned short*)(p); p += XSZ;
    unsigned short* xk   = (unsigned short*)(p); p += XSZ;
    unsigned short* xv   = (unsigned short*)(p); p += XSZ;
    unsigned short* qh   = (unsigned short*)(p); p += XSZ;
    unsigned short* kh   = (unsigned short*)(p); p += XSZ;
    unsigned short* vhT  = (unsigned short*)(p); p += XSZ;
    unsigned short* ctx  = (unsigned short*)(p); p += XSZ;
    float* xres = (float*)(p);                              // [4096,1024] f32

    // 1) weight transpose + bf16 convert
    {
        dim3 g(DD / 32, DD / 32), blk(32, 8);
        transpose_cvt_kernel<<<g, blk, 0, stream>>>(wq,  wqT,  DD, NN);
        transpose_cvt_kernel<<<g, blk, 0, stream>>>(wk,  wkT,  DD, NN);
        transpose_cvt_kernel<<<g, blk, 0, stream>>>(wv,  wvT,  DD, NN);
        transpose_cvt_kernel<<<g, blk, 0, stream>>>(wfc, wfcT, NN, DD);
    }
    // 2) activation bf16 convert
    {
        int n4 = (MM * KK) / 4;
        int blocks = n4 / 256;
        cvt_bf16_kernel<<<blocks, 256, 0, stream>>>(q, xq, n4);
        cvt_bf16_kernel<<<blocks, 256, 0, stream>>>(k, xk, n4);
        cvt_bf16_kernel<<<blocks, 256, 0, stream>>>(v, xv, n4);
    }
    // 3) QKV projections
    {
        dim3 g(NN / 64, MM / 128);
        gemm_bf16_kernel<<<g, 256, 0, stream>>>(xq, wqT, bq, nullptr, qh,  MM, NN, KK, 0);
        gemm_bf16_kernel<<<g, 256, 0, stream>>>(xk, wkT, bk, nullptr, kh,  MM, NN, KK, 0);
        gemm_bf16_kernel<<<g, 256, 0, stream>>>(xv, wvT, bv, nullptr, vhT, MM, NN, KK, 1);
    }
    // 4) scores + mask + softmax -> attn (f32, [H,B,L,L]) in d_out
    {
        dim3 g(LL / 16, HH, BB);
        attn_scores_softmax_kernel<<<g, 256, 0, stream>>>(qh, kh, mask, attn_out);
    }
    // 5) attn @ V -> ctx (bf16)
    {
        dim3 g(LL / 128, BB * HH);
        attn_av_kernel<<<g, 256, 0, stream>>>(attn_out, vhT, ctx);
    }
    // 6) FC + bias + residual -> xres (f32)
    {
        dim3 g(DD / 64, MM / 128);
        gemm_bf16_kernel<<<g, 256, 0, stream>>>(ctx, wfcT, bfc, q, xres, MM, DD, NN, 2);
    }
    // 7) LayerNorm -> y
    layernorm_kernel<<<MM / 8, 256, 0, stream>>>(xres, gamma, beta, y_out);
}
